// ISPGIN_30090540876437
// MI455X (gfx1250) — matrix operand order repaired and checked by reading, not compiled, not run on hardware
//
#include <hip/hip_runtime.h>
#include <hip/hip_bf16.h>

// ---------------------------------------------------------------------------
// ISPGIN on gfx1250: spectral GIN with WMMA f16 GEMMs, atomic scatter-add
// aggregation, host-side loop over K=8 frequencies.
// N = 100000 is an exact multiple of 16, so all 16-node WMMA tiles are full
// and no bounds guards are needed in the MLP kernels (keeps EXEC all-ones
// around v_wmma as the ISA requires, and keeps the store path branch-free).
// ---------------------------------------------------------------------------

#define N_NODES   100000
#define N_EDGES   3200000
#define N_GRAPHS  64
#define KFREQ     8
#define HDIM      64
#define NH        (N_NODES * HDIM)

typedef __attribute__((ext_vector_type(16))) _Float16 v16h;
typedef __attribute__((ext_vector_type(8)))  float    v8f;

union FragU { unsigned u[8]; v16h v; };

__device__ __forceinline__ unsigned short f2h(float x) {
  return __builtin_bit_cast(unsigned short, (_Float16)x);
}
__device__ __forceinline__ unsigned packh(float lo, float hi) {
  return (unsigned)f2h(lo) | ((unsigned)f2h(hi) << 16);
}
__device__ __forceinline__ v8f zero8() {
  v8f z;
#pragma unroll
  for (int i = 0; i < 8; ++i) z[i] = 0.0f;
  return z;
}

// A-fragment gather from LDS tile stored as u32[16][32] (each u32 = packed
// fp16 pair at k=2j, 2j+1). Layout per ISA 7.12.2 (16-bit A, 16x32):
// lanes 0-15: vgpr0-3 -> kk 0..7, vgpr4-7 -> kk 16..23
// lanes16-31: vgpr0-3 -> kk 8..15, vgpr4-7 -> kk 24..31
__device__ __forceinline__ void gather_a_frag(const unsigned* sA, int s,
                                              int lane, unsigned* out) {
  int m    = lane & 15;
  int hi16 = lane >> 4;
#pragma unroll
  for (int v = 0; v < 8; ++v) {
    int j = 16 * s + ((v < 4) ? (hi16 * 4 + v) : (8 + hi16 * 4 + (v - 4)));
    out[v] = sA[m * 32 + j];
  }
}

// B fragments prepacked in global: frag[(c*2+s)*256 + lane*8 + v]
__device__ __forceinline__ void load_b_frag(const unsigned* wf, int c, int s,
                                            int lane, unsigned* out) {
  const unsigned* p = wf + ((c * 2 + s) * 256 + lane * 8);
#pragma unroll
  for (int v = 0; v < 8; ++v) out[v] = p[v];
}

__device__ __forceinline__ v8f wmma_acc(const unsigned* au, const unsigned* bu,
                                        v8f acc) {
  FragU a, b;
#pragma unroll
  for (int i = 0; i < 8; ++i) { a.u[i] = au[i]; b.u[i] = bu[i]; }
  return __builtin_amdgcn_wmma_f32_16x16x32_f16(
      false, a.v, false, b.v, (short)0, acc, false, false);
}

// ---------------------------------------------------------------------------
// Weight prep: convert ten 64x64 f32 matrices (row = k/in, col = n/out) into
// fp16 WMMA B-fragment layout. One thread per packed u32 word (20480 total).
// B layout (16-bit B, 32x16): lane%16 = N col; lanes16-31 carry kk+8;
// vgpr v packs kk=2v (lo), 2v+1 (hi) for v<4, kk=16+2(v-4), +1 for v>=4.
// ---------------------------------------------------------------------------
__global__ void prep_weights(const float* __restrict__ w2r0,
                             const float* __restrict__ w2i0,
                             const float* __restrict__ w1r12,
                             const float* __restrict__ w2r12,
                             const float* __restrict__ w1i12,
                             const float* __restrict__ w2i12,
                             unsigned* __restrict__ frags) {
  int idx = blockIdx.x * 256 + threadIdx.x;
  if (idx >= 10 * 2048) return;
  int mm = idx >> 11;
  int w  = idx & 2047;
  const float* src;
  switch (mm) {
    case 0: src = w2r0; break;
    case 1: src = w2i0; break;
    case 2: src = w1r12; break;
    case 3: src = w1r12 + 4096; break;
    case 4: src = w2r12; break;
    case 5: src = w2r12 + 4096; break;
    case 6: src = w1i12; break;
    case 7: src = w1i12 + 4096; break;
    case 8: src = w2i12; break;
    default: src = w2i12 + 4096; break;
  }
  int cs   = w >> 8;          // (c*2 + s), 0..7
  int rem  = w & 255;         // lane*8 + v
  int lane = rem >> 3;
  int v    = rem & 7;
  int c = cs >> 1, s = cs & 1;
  int n   = c * 16 + (lane & 15);
  int kk0 = (lane >> 4) * 8 + 2 * v + ((v >= 4) ? 8 : 0);
  int k   = 32 * s + kk0;
  frags[idx] = packh(src[k * 64 + n], src[(k + 1) * 64 + n]);
}

__global__ void zero_f4(float* __restrict__ p, int nvec4) {
  int i = blockIdx.x * 256 + threadIdx.x;
  if (i < nvec4) {
    float4 z; z.x = z.y = z.z = z.w = 0.0f;
    ((float4*)p)[i] = z;
  }
}

// phase features for frequency k; also zero the scalar aggregation buffers
__global__ void init_freq(const float* __restrict__ rw, float omega,
                          float* __restrict__ zr0, float* __restrict__ zi0,
                          float* __restrict__ ar0, float* __restrict__ ai0,
                          int n) {
  int i = blockIdx.x * 256 + threadIdx.x;
  if (i >= n) return;
  float ph = omega * rw[i];
  zr0[i] = __cosf(ph);
  zi0[i] = __sinf(ph);
  ar0[i] = 0.0f;
  ai0[i] = 0.0f;
}

// layer-0 aggregation: scalar feature per node
__global__ void agg_scalar(const int* __restrict__ row,
                           const int* __restrict__ col,
                           const float* __restrict__ zr0,
                           const float* __restrict__ zi0,
                           float* __restrict__ ar0, float* __restrict__ ai0,
                           int e) {
  int i = blockIdx.x * 256 + threadIdx.x;
  if (i >= e) return;
  int r = row[i], c = col[i];
  atomicAdd(ar0 + c, zr0[r]);
  atomicAdd(ai0 + c, zi0[r]);
}

// layers 1-2 aggregation: H=64 features, float4 gather + 4 atomic adds each
__global__ void agg_vec(const int* __restrict__ row,
                        const int* __restrict__ col,
                        const float* __restrict__ Zr,
                        const float* __restrict__ Zi,
                        float* __restrict__ Ar, float* __restrict__ Ai,
                        int total) {
  int i = blockIdx.x * 256 + threadIdx.x;
  if (i >= total) return;
  int e = i >> 4;
  int q = (i & 15) * 4;
  int r = row[e], c = col[e];
  const float4 zr = *(const float4*)(Zr + r * 64 + q);
  const float4 zi = *(const float4*)(Zi + r * 64 + q);
  float* pr = Ar + c * 64 + q;
  float* pi = Ai + c * 64 + q;
  atomicAdd(pr + 0, zr.x); atomicAdd(pr + 1, zr.y);
  atomicAdd(pr + 2, zr.z); atomicAdd(pr + 3, zr.w);
  atomicAdd(pi + 0, zi.x); atomicAdd(pi + 1, zi.y);
  atomicAdd(pi + 2, zi.z); atomicAdd(pi + 3, zi.w);
}

// ---------------------------------------------------------------------------
// Layer 0 MLP: h = (1+eps)*z0 + a0 (scalar); t = relu(h*w1 + b1) (64-wide
// broadcast); Z = t @ w2 + b2 via WMMA. Block = 128 threads (4 waves); each
// wave owns one 16-col output tile of the 16-node row block.
// ---------------------------------------------------------------------------
__global__ void mlp_layer0(const float* __restrict__ z0,
                           const float* __restrict__ a0,
                           const float* __restrict__ w1,
                           const float* __restrict__ b1,
                           const unsigned* __restrict__ w2f,
                           const float* __restrict__ b2,
                           const float* __restrict__ epsArr,
                           float* __restrict__ Zout) {
  __shared__ unsigned sT[16 * 32];           // fp16-pair tile [16][64]
  const int tid  = threadIdx.x;
  const int tile = blockIdx.x;
  const float epl = 1.0f + epsArr[0];
  unsigned short* sTh = (unsigned short*)sT;

#pragma unroll
  for (int it = 0; it < 8; ++it) {
    int idx = tid + it * 128;
    int m = idx >> 6, h = idx & 63;
    int node = tile * 16 + m;
    float hv = epl * z0[node] + a0[node];
    float t  = hv * w1[h] + b1[h];
    sTh[idx] = f2h(t > 0.0f ? t : 0.0f);
  }
  __syncthreads();

  const int lane = tid & 31;
  const int wv   = tid >> 5;   // output column tile
  const int hi16 = lane >> 4;

  v8f acc = zero8();
#pragma unroll
  for (int s = 0; s < 2; ++s) {
    unsigned au[8], bu[8];
    gather_a_frag(sT, s, lane, au);
    load_b_frag(w2f, wv, s, lane, bu);
    acc = wmma_acc(au, bu, acc);
  }
  int n = wv * 16 + (lane & 15);
  float bias = b2[n];
#pragma unroll
  for (int r = 0; r < 8; ++r) {
    int node = tile * 16 + r + 8 * hi16;
    Zout[node * 64 + n] = acc[r] + bias;
  }
}

// ---------------------------------------------------------------------------
// Layers 1-2 MLP: A = (1+eps)*Z + Agg (16x64), relu(A@w1+b1)@w2+b2 via WMMA.
// ---------------------------------------------------------------------------
__global__ void mlp_hidden(const float* __restrict__ Zin,
                           const float* __restrict__ Agg,
                           const unsigned* __restrict__ w1f,
                           const float* __restrict__ b1,
                           const unsigned* __restrict__ w2f,
                           const float* __restrict__ b2,
                           const float* __restrict__ epsArr, int epsIdx,
                           float* __restrict__ Zout) {
  __shared__ unsigned sA[16 * 32];
  __shared__ unsigned sB[16 * 32];
  const int tid  = threadIdx.x;
  const int tile = blockIdx.x;
  const float epl = 1.0f + epsArr[epsIdx];

  // fused load: (1+eps)*Z + Agg, convert to packed fp16 pairs
#pragma unroll
  for (int it = 0; it < 4; ++it) {
    int w = tid + it * 128;
    int m = w >> 5, j = w & 31;
    int node = tile * 16 + m;
    int base = node * 64 + 2 * j;
    float lo = epl * Zin[base]     + Agg[base];
    float hi = epl * Zin[base + 1] + Agg[base + 1];
    sA[w] = packh(lo, hi);
  }
  __syncthreads();

  const int lane = tid & 31;
  const int wv   = tid >> 5;
  const int hi16 = lane >> 4;
  const int n    = wv * 16 + (lane & 15);

  // GEMM 1: A @ w1  (K = 64 -> two 16x16x32 WMMA steps)
  v8f acc = zero8();
#pragma unroll
  for (int s = 0; s < 2; ++s) {
    unsigned au[8], bu[8];
    gather_a_frag(sA, s, lane, au);
    load_b_frag(w1f, wv, s, lane, bu);
    acc = wmma_acc(au, bu, acc);
  }
  float bias1 = b1[n];
  unsigned short* sBh = (unsigned short*)sB;
#pragma unroll
  for (int r = 0; r < 8; ++r) {
    float t = acc[r] + bias1;
    t = t > 0.0f ? t : 0.0f;
    sBh[(r + 8 * hi16) * 64 + n] = f2h(t);
  }
  __syncthreads();

  // GEMM 2: relu(...) @ w2
  v8f acc2 = zero8();
#pragma unroll
  for (int s = 0; s < 2; ++s) {
    unsigned au[8], bu[8];
    gather_a_frag(sB, s, lane, au);
    load_b_frag(w2f, wv, s, lane, bu);
    acc2 = wmma_acc(au, bu, acc2);
  }
  float bias2 = b2[n];
#pragma unroll
  for (int r = 0; r < 8; ++r) {
    int node = tile * 16 + r + 8 * hi16;
    Zout[node * 64 + n] = acc2[r] + bias2;
  }
}

// magnitude + graph pooling for frequency k
__global__ void mag_pool(const float* __restrict__ Zr,
                         const float* __restrict__ Zi,
                         const int* __restrict__ batch,
                         float* __restrict__ xpool, int kbase, int n) {
  int i = blockIdx.x * 256 + threadIdx.x;
  if (i >= n * 16) return;
  int node = i >> 4;
  int q    = (i & 15) * 4;
  int g = batch[node];
  const float4 zr = *(const float4*)(Zr + node * 64 + q);
  const float4 zi = *(const float4*)(Zi + node * 64 + q);
  float4 m;
  m.x = sqrtf(zr.x * zr.x + zi.x * zi.x + 1e-8f);
  m.y = sqrtf(zr.y * zr.y + zi.y * zi.y + 1e-8f);
  m.z = sqrtf(zr.z * zr.z + zi.z * zi.z + 1e-8f);
  m.w = sqrtf(zr.w * zr.w + zi.w * zi.w + 1e-8f);
  float* p = xpool + g * (KFREQ * HDIM) + kbase + q;
  atomicAdd(p + 0, m.x); atomicAdd(p + 1, m.y);
  atomicAdd(p + 2, m.z); atomicAdd(p + 3, m.w);
}

// classifier stage 1: relu(x @ wc1 + bc1), x:(64,512) wc1:(512,64)
__global__ void cls_stage1(const float* __restrict__ xpool,
                           const float* __restrict__ wc1,
                           const float* __restrict__ bc1,
                           float* __restrict__ out1) {
  int idx = blockIdx.x * 256 + threadIdx.x;   // 4096 threads
  int g = idx >> 6, h = idx & 63;
  float s = bc1[h];
  for (int j = 0; j < 512; ++j) s += xpool[g * 512 + j] * wc1[j * 64 + h];
  out1[idx] = s > 0.0f ? s : 0.0f;
}

// classifier stage 2: out1 @ wc2 + bc2 -> (64, 2)
__global__ void cls_stage2(const float* __restrict__ out1,
                           const float* __restrict__ wc2,
                           const float* __restrict__ bc2,
                           float* __restrict__ out) {
  int idx = threadIdx.x;                      // 128 threads
  int g = idx >> 1, c = idx & 1;
  float s = bc2[c];
  for (int h = 0; h < 64; ++h) s += out1[g * 64 + h] * wc2[h * 2 + c];
  out[idx] = s;
}

extern "C" void kernel_launch(void* const* d_in, const int* in_sizes, int n_in,
                              void* d_out, int out_size, void* d_ws,
                              size_t ws_size, hipStream_t stream) {
  const float* rw    = (const float*)d_in[0];
  const int*   ei    = (const int*)d_in[1];     // (2, E): row then col
  const int*   batch = (const int*)d_in[2];
  const float* w1r0  = (const float*)d_in[4];
  const float* b1r0  = (const float*)d_in[5];
  const float* w2r0  = (const float*)d_in[6];
  const float* b2r0  = (const float*)d_in[7];
  const float* w1i0  = (const float*)d_in[8];
  const float* b1i0  = (const float*)d_in[9];
  const float* w2i0  = (const float*)d_in[10];
  const float* b2i0  = (const float*)d_in[11];
  const float* w1r12 = (const float*)d_in[12];
  const float* b1r12 = (const float*)d_in[13];
  const float* w2r12 = (const float*)d_in[14];
  const float* b2r12 = (const float*)d_in[15];
  const float* w1i12 = (const float*)d_in[16];
  const float* b1i12 = (const float*)d_in[17];
  const float* w2i12 = (const float*)d_in[18];
  const float* b2i12 = (const float*)d_in[19];
  const float* eps   = (const float*)d_in[20];
  const float* wc1   = (const float*)d_in[21];
  const float* bc1   = (const float*)d_in[22];
  const float* wc2   = (const float*)d_in[23];
  const float* bc2   = (const float*)d_in[24];

  const int N = N_NODES, E = N_EDGES;
  const int* row = ei;
  const int* col = ei + E;

  // workspace layout (floats)
  float* ws  = (float*)d_ws;
  float* zr0 = ws;                  // N
  float* zi0 = zr0 + N;             // N
  float* ar0 = zi0 + N;             // N
  float* ai0 = ar0 + N;             // N
  float* Zr  = ai0 + N;             // N*64
  float* Zi  = Zr + NH;
  float* Ar  = Zi + NH;             // Ar/Ai contiguous -> single zero pass
  float* Ai  = Ar + NH;
  float* Zr2 = Ai + NH;
  float* Zi2 = Zr2 + NH;
  float* xpool = Zi2 + NH;                        // 64*512
  float* out1  = xpool + N_GRAPHS * KFREQ * HDIM; // 64*64
  unsigned* frags = (unsigned*)(out1 + N_GRAPHS * HDIM);  // 10*2048 u32

  // one-time weight fragment prep (deterministic; rerun every call)
  prep_weights<<<80, 256, 0, stream>>>(w2r0, w2i0, w1r12, w2r12, w1i12, w2i12,
                                       frags);
  zero_f4<<<(8192 + 255) / 256, 256, 0, stream>>>(xpool, 8192);

  for (int k = 0; k < KFREQ; ++k) {
    float omega = 6.28318530717958647692f * (float)k / (float)KFREQ;
    init_freq<<<(N + 255) / 256, 256, 0, stream>>>(rw, omega, zr0, zi0, ar0,
                                                   ai0, N);
    agg_scalar<<<(E + 255) / 256, 256, 0, stream>>>(row, col, zr0, zi0, ar0,
                                                    ai0, E);
    mlp_layer0<<<N / 16, 128, 0, stream>>>(zr0, ar0, w1r0, b1r0,
                                           frags + 0 * 2048, b2r0, eps, Zr);
    mlp_layer0<<<N / 16, 128, 0, stream>>>(zi0, ai0, w1i0, b1i0,
                                           frags + 1 * 2048, b2i0, eps, Zi);

    for (int l = 0; l < 2; ++l) {
      float* ZinR = l ? Zr2 : Zr;
      float* ZinI = l ? Zi2 : Zi;
      float* ZoR  = l ? Zr : Zr2;
      float* ZoI  = l ? Zi : Zi2;
      zero_f4<<<(2 * NH / 4 + 255) / 256, 256, 0, stream>>>(Ar, 2 * NH / 4);
      agg_vec<<<(E * 16 + 255) / 256, 256, 0, stream>>>(row, col, ZinR, ZinI,
                                                        Ar, Ai, E * 16);
      mlp_hidden<<<N / 16, 128, 0, stream>>>(
          ZinR, Ar, frags + (2 + l) * 2048, b1r12 + l * 64,
          frags + (4 + l) * 2048, b2r12 + l * 64, eps, l + 1, ZoR);
      mlp_hidden<<<N / 16, 128, 0, stream>>>(
          ZinI, Ai, frags + (6 + l) * 2048, b1i12 + l * 64,
          frags + (8 + l) * 2048, b2i12 + l * 64, eps, l + 1, ZoI);
    }
    // after l-loop the final features sit in Zr / Zi
    mag_pool<<<(N * 16 + 255) / 256, 256, 0, stream>>>(Zr, Zi, batch, xpool,
                                                       k * HDIM, N);
  }

  cls_stage1<<<16, 256, 0, stream>>>(xpool, wc1, bc1, out1);
  cls_stage2<<<1, 128, 0, stream>>>(out1, wc2, bc2, (float*)d_out);
}
